// SelfAttention_78915729097032
// MI455X (gfx1250) — compile-verified, matrix operand order checked by reading, not run on hardware
//
#include <hip/hip_runtime.h>
#include <math.h>

// ---------------------------------------------------------------------------
// E(3)-equivariant edge attention for MI455X (gfx1250, wave32, WMMA).
// Radial MLPs (~37 GFLOP) run on v_wmma_f32_16x16x32_f16; LDS staging is
// bank-conflict-free (padded strides); tp phase uses float4 LDS reads.
// ---------------------------------------------------------------------------

typedef __attribute__((ext_vector_type(16))) _Float16 v16h;
typedef __attribute__((ext_vector_type(8)))  float    v8f;

#define N_NODES   30000
#define E_EDGES   480000
#define MUL0_IN   16
#define MUL1_IN   8
#define MUL0_OUT  8
#define MUL1_OUT  4
#define NB        16
#define NH        64
#define WNUM      288          // 128 + 64 + 64 + 32
#define XDIM      40           // 16 scalars + 8*3 vectors
#define QDIM      20           // q0(8) + q1(12)
#define VDIM      20           // v0(8) + v1(12)

#define F2_NTILES 18           // 288 / 16
#define FRAG_HALF 512          // one 32x16 B fragment: 32 lanes x 16 halfs
#define NFRAGS    (4 + 36 + 4 + 36)

// LDS strides padded for conflict-free access (64 banks x 4B):
//  - H_STRIDE = 72 halfs = 36 dwords: m*36 mod 64 distinct for m=0..15, 16B-aligned
//  - W_STRIDE = 292 floats: m*36 mod 64 distinct (292 mod 64 = 36), 16B-aligned
#define H_STRIDE  72
#define W_STRIDE  292

#define WAVES_PER_BLOCK 2
#define TILE_E          16

// scaling constants folded from the reference
#define C_SS   0.17677669529663687f   // 1/(4*sqrt2)
#define C_UU   0.14433756729740643f   // 1/sqrt(48)
#define C_SU   0.17677669529663687f   // 1/(4*sqrt2)
#define C_US   0.25f                  // 1/(sqrt8*sqrt2)
#define C_D0   0.08838834764831845f   // 1/(8*sqrt2)
#define C_D1   0.10206207261596577f   // 1/(4*sqrt6)
#define C_Q0   0.25f                  // 1/sqrt(16)
#define C_Q1   0.35355339059327379f   // 1/sqrt(8)
#define MAX_RADIUS 3.15f

// ---------------------------------------------------------------------------
__global__ void zero_kernel(float* __restrict__ Z, float* __restrict__ out) {
  int idx = blockIdx.x * blockDim.x + threadIdx.x;
  if (idx < N_NODES * QDIM) out[idx] = 0.0f;
  if (idx < N_NODES)        Z[idx]   = 0.0f;
}

// ---------------------------------------------------------------------------
// Pre-swizzle W{k,v}{1,2} (scaled) into WMMA B-fragment layout.
// B fragment (32x16, f16): lane n16 = lane&15 owns column n; the two lane
// halves split K like the A-matrix layout:
//   lanes 0-15 : t=0..7 -> K = KB+0..7,  t=8..15 -> K = KB+16..23
//   lanes 16-31: t=0..7 -> K = KB+8..15, t=8..15 -> K = KB+24..31
// Each lane's 16 halfs are contiguous (32B) -> two global_load_b128 per frag.
// ---------------------------------------------------------------------------
__global__ void prep_frags(const float* __restrict__ Wk1, const float* __restrict__ Wk2,
                           const float* __restrict__ Wv1, const float* __restrict__ Wv2,
                           _Float16* __restrict__ frags) {
  for (int idx = blockIdx.x * blockDim.x + threadIdx.x;
       idx < NFRAGS * FRAG_HALF; idx += blockDim.x * gridDim.x) {
    int frag   = idx >> 9;
    int within = idx & 511;
    int lane   = within >> 4;
    int t      = within & 15;
    int klocal = ((t < 8) ? t : t + 8) + ((lane >= 16) ? 8 : 0);  // 0..31
    int n16    = lane & 15;
    float val = 0.0f;
    if (frag < 4) {                       // layer-1 K: 32x64, rows >=16 are pad
      int n = frag * 16 + n16;
      if (klocal < NB) val = Wk1[klocal * NH + n] * C_Q0;     // /sqrt(NB)=1/4
    } else if (frag < 40) {               // layer-2 K: 64x288
      int f = frag - 4, kk = f / F2_NTILES, nt = f % F2_NTILES;
      val = Wk2[(kk * 32 + klocal) * WNUM + nt * 16 + n16] * 0.125f; // /sqrt(NH)
    } else if (frag < 44) {               // layer-1 V
      int n = (frag - 40) * 16 + n16;
      if (klocal < NB) val = Wv1[klocal * NH + n] * C_Q0;
    } else {                              // layer-2 V
      int f = frag - 44, kk = f / F2_NTILES, nt = f % F2_NTILES;
      val = Wv2[(kk * 32 + klocal) * WNUM + nt * 16 + n16] * 0.125f;
    }
    frags[idx] = (_Float16)val;
  }
}

// ---------------------------------------------------------------------------
// Per-node queries: q0 = s @ Wq0/4 ; q1[b,c] = sum_a u[a,c] Wq1[a,b] / sqrt8.
// ---------------------------------------------------------------------------
__global__ void node_q_kernel(const float* __restrict__ x,
                              const float* __restrict__ Wq0,
                              const float* __restrict__ Wq1,
                              float* __restrict__ qbuf) {
  int n = blockIdx.x * blockDim.x + threadIdx.x;
  if (n >= N_NODES) return;
  const float* xr = x + (size_t)n * XDIM;
  float* q = qbuf + (size_t)n * QDIM;
  #pragma unroll
  for (int b = 0; b < MUL0_OUT; ++b) {
    float acc = 0.0f;
    #pragma unroll
    for (int a = 0; a < MUL0_IN; ++a) acc += xr[a] * Wq0[a * MUL0_OUT + b];
    q[b] = acc * C_Q0;
  }
  #pragma unroll
  for (int b = 0; b < MUL1_OUT; ++b) {
    float acc[3] = {0.f, 0.f, 0.f};
    #pragma unroll
    for (int a = 0; a < MUL1_IN; ++a) {
      float w = Wq1[a * MUL1_OUT + b];
      acc[0] += xr[16 + a * 3 + 0] * w;
      acc[1] += xr[16 + a * 3 + 1] * w;
      acc[2] += xr[16 + a * 3 + 2] * w;
    }
    q[8 + b * 3 + 0] = acc[0] * C_Q1;
    q[8 + b * 3 + 1] = acc[1] * C_Q1;
    q[8 + b * 3 + 2] = acc[2] * C_Q1;
  }
}

// ---------------------------------------------------------------------------
// One radial MLP for a 16-edge tile (one wave): 4 + 36 WMMAs (dynamic).
// hbuf: 16 x H_STRIDE f16 LDS stage (C-layout writes -> A-layout reads;
//       same-wave DS ops are in-order, compiler inserts s_wait_dscnt).
// wbuf: 16 x W_STRIDE f32 result in LDS.
// ---------------------------------------------------------------------------
__device__ __forceinline__ void radial_gemm(const v16h a1,
                                            const _Float16* __restrict__ frag1,
                                            const _Float16* __restrict__ frag2,
                                            _Float16* hbuf, float* wbuf, int lane) {
  const int m16  = lane & 15;
  const int mofs = (lane < 16) ? 0 : 8;
  // ---- layer 1: [16x32] x [32x64] + silu -> hbuf
  #pragma unroll
  for (int nt = 0; nt < 4; ++nt) {
    v16h b = *(const v16h*)(frag1 + nt * FRAG_HALF + lane * 16);
    v8f  c = {};
    c = __builtin_amdgcn_wmma_f32_16x16x32_f16(false, a1, false, b,
                                               (short)0, c, false, false);
    #pragma unroll
    for (int r = 0; r < 8; ++r) {
      float v  = c[r];
      float sl = v * __frcp_rn(1.0f + __expf(-v));          // silu, hw exp
      hbuf[(r + mofs) * H_STRIDE + nt * 16 + m16] = (_Float16)sl;
    }
  }
  // ---- restripe h into two layer-2 A fragments (K=64 -> 2 x 16x32)
  v16h a2[2];
  #pragma unroll
  for (int kk = 0; kk < 2; ++kk) {
    int kbase = kk * 32 + ((lane < 16) ? 0 : 8);
    #pragma unroll
    for (int t = 0; t < 8; ++t) a2[kk][t]     = hbuf[m16 * H_STRIDE + kbase + t];
    #pragma unroll
    for (int t = 0; t < 8; ++t) a2[kk][8 + t] = hbuf[m16 * H_STRIDE + kbase + 16 + t];
  }
  // ---- layer 2: [16x64] x [64x288] -> wbuf
  // unroll capped at 6 to limit live B fragments / VGPR pressure
  #pragma unroll 6
  for (int nt = 0; nt < F2_NTILES; ++nt) {
    v8f acc = {};
    v16h b0 = *(const v16h*)(frag2 + (0 * F2_NTILES + nt) * FRAG_HALF + lane * 16);
    acc = __builtin_amdgcn_wmma_f32_16x16x32_f16(false, a2[0], false, b0,
                                                 (short)0, acc, false, false);
    v16h b1 = *(const v16h*)(frag2 + (1 * F2_NTILES + nt) * FRAG_HALF + lane * 16);
    acc = __builtin_amdgcn_wmma_f32_16x16x32_f16(false, a2[1], false, b1,
                                                 (short)0, acc, false, false);
    #pragma unroll
    for (int r = 0; r < 8; ++r)
      wbuf[(r + mofs) * W_STRIDE + nt * 16 + m16] = acc[r];
  }
}

// ---------------------------------------------------------------------------
// Tensor product of one edge, accumulator form with float4 LDS reads
// (2 x ds_load_b128 per 'a' instead of 8 scalar loads).
//   out0[b] = sum_a xsy[a]*w[a*8+b] + sum_a uyc[a]*w[128+a*8+b]
//   out1[b,c] = tmp[b]*y1[c] + us[c][b]
// with scale factors folded into the per-a multipliers.
// ---------------------------------------------------------------------------
__device__ __forceinline__ void tp_compute(const float* __restrict__ wrow,
                                           const float* __restrict__ xs,
                                           const float* __restrict__ xu, // [8*3]
                                           float y0, const float* y1,
                                           float* out0, float* out1) {
  float o0[8] = {0.f,0.f,0.f,0.f,0.f,0.f,0.f,0.f};
  // ss block: multiplier xs[a]*y0*C_SS
  #pragma unroll
  for (int a = 0; a < MUL0_IN; ++a) {
    float m = xs[a] * y0 * C_SS;
    float4 lo = *(const float4*)(wrow + a * 8);
    float4 hi = *(const float4*)(wrow + a * 8 + 4);
    o0[0] += m * lo.x; o0[1] += m * lo.y; o0[2] += m * lo.z; o0[3] += m * lo.w;
    o0[4] += m * hi.x; o0[5] += m * hi.y; o0[6] += m * hi.z; o0[7] += m * hi.w;
  }
  // uu block: multiplier (xu[a]·y1)*C_UU
  #pragma unroll
  for (int a = 0; a < MUL1_IN; ++a) {
    float m = (xu[a*3+0]*y1[0] + xu[a*3+1]*y1[1] + xu[a*3+2]*y1[2]) * C_UU;
    float4 lo = *(const float4*)(wrow + 128 + a * 8);
    float4 hi = *(const float4*)(wrow + 128 + a * 8 + 4);
    o0[0] += m * lo.x; o0[1] += m * lo.y; o0[2] += m * lo.z; o0[3] += m * lo.w;
    o0[4] += m * hi.x; o0[5] += m * hi.y; o0[6] += m * hi.z; o0[7] += m * hi.w;
  }
  #pragma unroll
  for (int b = 0; b < 8; ++b) out0[b] = o0[b];

  // su block: tmp[b] = sum_a xs[a]*C_SU * w[192+a*4+b]
  float4 tmp = {0.f,0.f,0.f,0.f};
  #pragma unroll
  for (int a = 0; a < MUL0_IN; ++a) {
    float m = xs[a] * C_SU;
    float4 w4 = *(const float4*)(wrow + 192 + a * 4);
    tmp.x += m * w4.x; tmp.y += m * w4.y; tmp.z += m * w4.z; tmp.w += m * w4.w;
  }
  // us block: us[c][b] = sum_a xu[a,c]*y0*C_US * w[256+a*4+b]
  float4 us0 = {0.f,0.f,0.f,0.f}, us1 = {0.f,0.f,0.f,0.f}, us2 = {0.f,0.f,0.f,0.f};
  #pragma unroll
  for (int a = 0; a < MUL1_IN; ++a) {
    float4 w4 = *(const float4*)(wrow + 256 + a * 4);
    float m0 = xu[a*3+0] * y0 * C_US;
    float m1 = xu[a*3+1] * y0 * C_US;
    float m2 = xu[a*3+2] * y0 * C_US;
    us0.x += m0*w4.x; us0.y += m0*w4.y; us0.z += m0*w4.z; us0.w += m0*w4.w;
    us1.x += m1*w4.x; us1.y += m1*w4.y; us1.z += m1*w4.z; us1.w += m1*w4.w;
    us2.x += m2*w4.x; us2.y += m2*w4.y; us2.z += m2*w4.z; us2.w += m2*w4.w;
  }
  const float t4[4] = {tmp.x, tmp.y, tmp.z, tmp.w};
  const float u0[4] = {us0.x, us0.y, us0.z, us0.w};
  const float u1[4] = {us1.x, us1.y, us1.z, us1.w};
  const float u2[4] = {us2.x, us2.y, us2.z, us2.w};
  #pragma unroll
  for (int b = 0; b < MUL1_OUT; ++b) {
    out1[b*3+0] = t4[b] * y1[0] + u0[b];
    out1[b*3+1] = t4[b] * y1[1] + u1[b];
    out1[b*3+2] = t4[b] * y1[2] + u2[b];
  }
}

// ---------------------------------------------------------------------------
// Pass 1: one wave per 16-edge tile. Radial K -> tp -> logit d -> e, Z;
// radial V -> tp -> vflat.
// ---------------------------------------------------------------------------
__global__ __launch_bounds__(WAVES_PER_BLOCK * 32)
void edge_pass1(const float* __restrict__ x,   const int* __restrict__ eidx,
                const float* __restrict__ eattr, const float* __restrict__ eemb,
                const float* __restrict__ elen,
                const float* __restrict__ Wd0, const float* __restrict__ Wd1,
                const _Float16* __restrict__ frags,
                const float* __restrict__ qbuf,
                float* __restrict__ Z, float* __restrict__ ebuf,
                float* __restrict__ vflat) {
  __shared__ __align__(16) _Float16 lds_h[WAVES_PER_BLOCK][16 * H_STRIDE]; // 2.25 KB/wave
  __shared__ __align__(16) float    lds_w[WAVES_PER_BLOCK][16 * W_STRIDE]; // 18.25 KB/wave

  const int lane = threadIdx.x & 31;
  const int wv   = threadIdx.x >> 5;
  const long tile = (long)blockIdx.x * WAVES_PER_BLOCK + wv;
  const long e0   = tile * TILE_E;
  if (e0 >= E_EDGES) return;

  const _Float16* fragK1 = frags;
  const _Float16* fragK2 = frags + 4 * FRAG_HALF;
  const _Float16* fragV1 = frags + 40 * FRAG_HALF;
  const _Float16* fragV2 = frags + 44 * FRAG_HALF;

  // --- build layer-1 A fragment from edge_len_embbed (K padded 16->32)
  const int  m16  = lane & 15;
  const long arow = e0 + m16;
  const int  kb   = (lane < 16) ? 0 : 8;
  v16h a1;
  #pragma unroll
  for (int t = 0; t < 16; ++t) a1[t] = (_Float16)0.0f;
  if (arow < E_EDGES) {
    const float* src = eemb + arow * NB + kb;
    #pragma unroll
    for (int t = 0; t < 8; ++t) a1[t] = (_Float16)src[t];
    // prefetch next tile's embedding rows (global_prefetch_b8)
    if (arow + TILE_E * WAVES_PER_BLOCK < E_EDGES)
      __builtin_prefetch(eemb + (arow + TILE_E * WAVES_PER_BLOCK) * NB + kb, 0, 0);
  }

  // per-lane edge for the tp/logit phase (duplicated across the two halves)
  const long edge = e0 + m16;
  const int  ii = eidx[edge];
  const int  jj = eidx[E_EDGES + edge];
  const float y0 = eattr[edge * 4 + 0];
  float y1[3] = {eattr[edge*4+1], eattr[edge*4+2], eattr[edge*4+3]};
  float xs[MUL0_IN], xu[MUL1_IN * 3];
  {
    const float* xr = x + (size_t)ii * XDIM;
    #pragma unroll
    for (int a = 0; a < MUL0_IN; ++a) xs[a] = xr[a];
    #pragma unroll
    for (int a = 0; a < MUL1_IN * 3; ++a) xu[a] = xr[16 + a];
  }

  // ================= K radial + tensor product + logit =================
  radial_gemm(a1, fragK1, fragK2, lds_h[wv], lds_w[wv], lane);
  float k0[MUL0_OUT], k1[MUL1_OUT * 3];
  tp_compute(&lds_w[wv][m16 * W_STRIDE], xs, xu, y0, y1, k0, k1);

  const float* q = qbuf + (size_t)jj * QDIM;
  float dq0 = 0.f;
  #pragma unroll
  for (int a = 0; a < MUL0_OUT; ++a) {
    float qa = q[a];
    #pragma unroll
    for (int b = 0; b < MUL0_OUT; ++b) dq0 += qa * k0[b] * Wd0[a * MUL0_OUT + b];
  }
  float dq1 = 0.f;
  #pragma unroll
  for (int a = 0; a < MUL1_OUT; ++a) {
    #pragma unroll
    for (int b = 0; b < MUL1_OUT; ++b) {
      float s = q[8 + a*3 + 0] * k1[b*3 + 0]
              + q[8 + a*3 + 1] * k1[b*3 + 1]
              + q[8 + a*3 + 2] * k1[b*3 + 2];
      dq1 += s * Wd1[a * MUL1_OUT + b];
    }
  }
  const float d = dq0 * C_D0 + dq1 * C_D1;
  const float tcut = 10.0f * (1.0f - elen[edge] / MAX_RADIUS);
  const float cutoff = (tcut > 0.0f) ? __expf(-1.0f / fmaxf(tcut, 1e-6f)) : 0.0f;
  const float ee = cutoff * __expf(d);
  if (lane < 16) {
    ebuf[edge] = ee;
    __hip_atomic_fetch_add(&Z[jj], ee, __ATOMIC_RELAXED, __HIP_MEMORY_SCOPE_AGENT);
  }

  // ================= V radial + tensor product -> vflat ================
  radial_gemm(a1, fragV1, fragV2, lds_h[wv], lds_w[wv], lane);
  float v0[MUL0_OUT], v1[MUL1_OUT * 3];
  tp_compute(&lds_w[wv][m16 * W_STRIDE], xs, xu, y0, y1, v0, v1);
  if (lane < 16) {
    float* vo = vflat + edge * VDIM;
    #pragma unroll
    for (int b = 0; b < MUL0_OUT; ++b) vo[b] = v0[b];
    #pragma unroll
    for (int b = 0; b < MUL1_OUT * 3; ++b) vo[8 + b] = v1[b];
  }
}

// ---------------------------------------------------------------------------
// Pass 2: alpha = e/Z[j]; out[j] += sqrt(relu(alpha)) * vflat.
// ---------------------------------------------------------------------------
__global__ void scatter_kernel(const int* __restrict__ eidx,
                               const float* __restrict__ ebuf,
                               const float* __restrict__ Z,
                               const float* __restrict__ vflat,
                               float* __restrict__ out) {
  long e = (long)blockIdx.x * blockDim.x + threadIdx.x;
  if (e >= E_EDGES) return;
  const int jj = eidx[E_EDGES + e];
  const float alpha = ebuf[e] / Z[jj];
  const float wgt = sqrtf(fmaxf(alpha, 0.0f));
  const float* v = vflat + e * VDIM;
  float* o = out + (size_t)jj * VDIM;
  #pragma unroll
  for (int c = 0; c < VDIM; ++c)
    __hip_atomic_fetch_add(o + c, wgt * v[c],
                           __ATOMIC_RELAXED, __HIP_MEMORY_SCOPE_AGENT);
}

// ---------------------------------------------------------------------------
extern "C" void kernel_launch(void* const* d_in, const int* in_sizes, int n_in,
                              void* d_out, int out_size, void* d_ws, size_t ws_size,
                              hipStream_t stream) {
  const float* x     = (const float*)d_in[0];
  const int*   eidx  = (const int*)  d_in[1];
  const float* eattr = (const float*)d_in[2];
  const float* eemb  = (const float*)d_in[3];
  const float* elen  = (const float*)d_in[4];
  const float* Wq0   = (const float*)d_in[5];
  const float* Wq1   = (const float*)d_in[6];
  const float* Wk1   = (const float*)d_in[7];
  const float* Wk2   = (const float*)d_in[8];
  const float* Wv1   = (const float*)d_in[9];
  const float* Wv2   = (const float*)d_in[10];
  const float* Wd0   = (const float*)d_in[11];
  const float* Wd1   = (const float*)d_in[12];
  float* out = (float*)d_out;

  // workspace layout (~43 MB)
  float* ws    = (float*)d_ws;
  float* qbuf  = ws;                                    // N*20
  float* Zbuf  = qbuf + (size_t)N_NODES * QDIM;         // N
  float* ebuf  = Zbuf + N_NODES;                        // E
  float* vbuf  = ebuf + E_EDGES;                        // E*20
  _Float16* frags = (_Float16*)(vbuf + (size_t)E_EDGES * VDIM); // 80 KB

  zero_kernel<<<(N_NODES * QDIM + 255) / 256, 256, 0, stream>>>(Zbuf, out);
  prep_frags<<<(NFRAGS * FRAG_HALF + 255) / 256, 256, 0, stream>>>(
      Wk1, Wk2, Wv1, Wv2, frags);
  node_q_kernel<<<(N_NODES + 255) / 256, 256, 0, stream>>>(x, Wq0, Wq1, qbuf);

  const int tiles  = E_EDGES / TILE_E;                  // 30000
  const int blocks = (tiles + WAVES_PER_BLOCK - 1) / WAVES_PER_BLOCK;
  edge_pass1<<<blocks, WAVES_PER_BLOCK * 32, 0, stream>>>(
      x, eidx, eattr, eemb, elen, Wd0, Wd1, frags, qbuf, Zbuf, ebuf, vbuf);

  scatter_kernel<<<(E_EDGES + 255) / 256, 256, 0, stream>>>(
      eidx, ebuf, Zbuf, vbuf, out);
}
